// VarianceAdaptor_22625887715387
// MI455X (gfx1250) — compile-verified, matrix operand order checked by reading, not run on hardware
//
#include <hip/hip_runtime.h>
#include <hip/hip_bf16.h>

// ---------------------------------------------------------------------------
// Problem constants (from reference)
// ---------------------------------------------------------------------------
#define BB 64
#define TT 256
#define MM 1280
#define DD 384
#define FF 384
#define NBINS 256
#define LN_EPS 1e-5f

typedef __attribute__((ext_vector_type(16))) __bf16        bf16x16;
typedef __attribute__((ext_vector_type(8)))  float         f32x8;
typedef __attribute__((ext_vector_type(4)))  unsigned int  u32x4;

union FragAB { bf16x16 v; u32x4 h[2]; };

// round-to-nearest-even f32 -> bf16 (as raw u16)
__device__ __forceinline__ unsigned short f2bf(float f) {
    unsigned int u = __builtin_bit_cast(unsigned int, f);
    u += 0x7FFFu + ((u >> 16) & 1u);
    return (unsigned short)(u >> 16);
}

// ---------------------------------------------------------------------------
// zero the two halo rows (t=-1, t=L) of a padded bf16 buffer [B][L+2][DD]
// ---------------------------------------------------------------------------
__global__ void zero_pad_rows_kernel(unsigned short* __restrict__ buf, int L) {
    const int b    = blockIdx.x >> 1;
    const int side = blockIdx.x & 1;
    const long long prow = (long long)b * (L + 2) + (side ? (L + 1) : 0);
    unsigned short* p = buf + prow * DD;
    for (int d = threadIdx.x; d < DD; d += blockDim.x) p[d] = 0;
}

// ---------------------------------------------------------------------------
// f32 [B][T][D] -> padded bf16 [B][T+2][D] (data rows only; halos zeroed above)
// ---------------------------------------------------------------------------
__global__ void f32_to_bf16_pad_kernel(const float* __restrict__ in,
                                       unsigned short* __restrict__ out,
                                       int L, long long n) {
    long long i = (long long)blockIdx.x * blockDim.x + threadIdx.x;
    if (i >= n) return;                         // n = B*L*DD
    const long long b   = i / ((long long)L * DD);
    const long long rem = i - b * (long long)L * DD;
    out[(b * (L + 2) + 1) * DD + rem] = f2bf(in[i]);
}

// ---------------------------------------------------------------------------
// weight prep: w[k][cin][cout] (f32) -> wT[k][cout][cin] (bf16)
// ---------------------------------------------------------------------------
__global__ void wprep_kernel(const float* __restrict__ w,
                             unsigned short* __restrict__ wT) {
    int o = blockIdx.x * blockDim.x + threadIdx.x;       // [k][cout][cin] flat
    if (o >= 3 * DD * FF) return;
    int k    = o / (DD * FF);
    int r    = o - k * DD * FF;
    int cout = r / DD;
    int cin  = r - cout * DD;
    wT[o] = f2bf(w[(k * DD + cin) * FF + cout]);
}

// ---------------------------------------------------------------------------
// duration cumsum (one block per batch, T=256 threads) + mel_len output
// ---------------------------------------------------------------------------
__global__ void dur_scan_kernel(const int* __restrict__ dur,
                                int* __restrict__ ends,
                                float* __restrict__ mel_len_out) {
    __shared__ int sh[TT];
    int b = blockIdx.x, t = threadIdx.x;
    sh[t] = dur[b * TT + t];
    __syncthreads();
    for (int off = 1; off < TT; off <<= 1) {
        int add = (t >= off) ? sh[t - off] : 0;
        __syncthreads();
        sh[t] += add;
        __syncthreads();
    }
    ends[b * TT + t] = sh[t];
    if (t == TT - 1) {
        int ml = sh[t] < MM ? sh[t] : MM;
        mel_len_out[b] = (float)ml;
    }
}

// ---------------------------------------------------------------------------
// length regulate: xe[b,m,:] = x[b, searchsorted_right(ends, m), :]
// one block (128 thr) per (b, m) frame; f32 flat copy + bf16 padded copy
// ---------------------------------------------------------------------------
__global__ void length_reg_kernel(const float* __restrict__ x,
                                  const int* __restrict__ ends,
                                  float* __restrict__ xe,
                                  unsigned short* __restrict__ xebf_pad) {
    int b = blockIdx.x / MM;
    int m = blockIdx.x - b * MM;
    const int* e = ends + b * TT;
    int lo = 0, hi = TT;
    while (lo < hi) { int mid = (lo + hi) >> 1; if (e[mid] <= m) lo = mid + 1; else hi = mid; }
    bool valid = lo < TT;
    int ic = valid ? lo : TT - 1;
    long long src  = ((long long)b * TT + ic) * DD;
    long long dst  = ((long long)b * MM + m) * DD;
    long long dstp = ((long long)b * (MM + 2) + 1 + m) * DD;
    for (int d = threadIdx.x; d < DD; d += blockDim.x) {
        float v = valid ? x[src + d] : 0.0f;
        xe[dst + d]        = v;
        xebf_pad[dstp + d] = f2bf(v);
    }
}

// ---------------------------------------------------------------------------
// Conv1d(K=3) as GEMM with K' = 3*384, bf16 WMMA, f32 accumulate.
//   Apad : bf16 activations [B][L+2][384] (zero halo rows; L | 128)
//   Wt   : bf16 weights     [3][384 cout][384 cin]
//   H    : f32 out          [B*L][384], bias + ReLU fused
//
// Workgroup: 8 waves -> 128 rows x 192 cols; grid = (nRows/128, 2 col groups).
// Wave tile: 2 row tiles x 6 col tiles (32 rows x 96 cols, 12 accumulators).
// Zero-padded A makes every tap load unconditional, so ALL fragment loads for
// all 36 K-steps are immediate offsets off two base pointers. Software
// pipelining with prefetch distance 1 keeps a full K-step (8 b128-pairs) in
// flight behind each 12-WMMA chain.
// ---------------------------------------------------------------------------
__global__ __launch_bounds__(256)
void conv_gemm_kernel(const unsigned short* __restrict__ Apad,
                      const unsigned short* __restrict__ Wt,
                      const float* __restrict__ bias,
                      float* __restrict__ H,
                      int L) {
    const int tid  = threadIdx.x;
    const int wave = tid >> 5;
    const int lane = tid & 31;
    const int wr   = wave & 3;          // row group within block (32 rows each)
    const int wc   = wave >> 2;         // col group within block (96 cols each)
    const int r0   = blockIdx.x * 128 + wr * 32;
    const int colbase = blockIdx.y * 192 + wc * 96;
    const int lm   = lane & 15;
    const int hs   = lane >> 4;         // K half select
    const int koff = hs * 8;

    const int row = r0 + lm;            // first row tile's row for this lane
    const int b   = row / L;
    const int t   = row - b * L;

    f32x8 acc[2][6];
#pragma unroll
    for (int i = 0; i < 2; ++i)
#pragma unroll
        for (int c = 0; c < 6; ++c) acc[i][c] = 0.0f;

    // single base pointer for A (row tile 1 at constant +16*DD), one for B
    const unsigned short* pa = Apad + ((long long)b * (L + 2) + t) * DD + koff;
    const unsigned short* pb = Wt + (long long)(colbase + lm) * DD + koff;

    struct Step { FragAB a0, a1, b[6]; };
    auto loadStep = [&](int kidx, Step& s) {
        const int tap = kidx / 12;
        const int c0  = (kidx - tap * 12) * 32;
        const int ao  = tap * DD + c0;          // A: tap shift + cin block
        const int bo  = tap * FF * DD + c0;     // B: tap plane + cin block
        s.a0.h[0] = *(const u32x4*)(pa + ao);
        s.a0.h[1] = *(const u32x4*)(pa + ao + 16);
        s.a1.h[0] = *(const u32x4*)(pa + ao + 16 * DD);
        s.a1.h[1] = *(const u32x4*)(pa + ao + 16 * DD + 16);
#pragma unroll
        for (int ct = 0; ct < 6; ++ct) {
            s.b[ct].h[0] = *(const u32x4*)(pb + bo + ct * 16 * DD);
            s.b[ct].h[1] = *(const u32x4*)(pb + bo + ct * 16 * DD + 16);
        }
    };

    Step cur, nxt;
    loadStep(0, cur);
#pragma unroll
    for (int k = 0; k < 36; ++k) {
        if (k < 35) loadStep(k + 1, nxt);       // prefetch next K-step
#pragma unroll
        for (int ct = 0; ct < 6; ++ct) {
            acc[0][ct] = __builtin_amdgcn_wmma_f32_16x16x32_bf16(
                false, cur.a0.v, false, cur.b[ct].v, (short)0, acc[0][ct], false, false);
            acc[1][ct] = __builtin_amdgcn_wmma_f32_16x16x32_bf16(
                false, cur.a1.v, false, cur.b[ct].v, (short)0, acc[1][ct], false, false);
        }
        cur = nxt;                              // renamed away after unroll
    }

    // epilogue: bias + ReLU, f32 store (H is flat [B*L][FF])
#pragma unroll
    for (int ct = 0; ct < 6; ++ct) {
        const int col = colbase + ct * 16 + lm;
        const float bv = bias[col];
#pragma unroll
        for (int i = 0; i < 2; ++i) {
#pragma unroll
            for (int k = 0; k < 8; ++k) {
                const int orow = r0 + i * 16 + k + hs * 8;
                float v = acc[i][ct][k] + bv;
                v = v > 0.0f ? v : 0.0f;
                H[(long long)orow * FF + col] = v;
            }
        }
    }
}

// ---------------------------------------------------------------------------
// LayerNorm over F=384, wave-per-row, bf16 output into padded layout
// ---------------------------------------------------------------------------
__global__ __launch_bounds__(256)
void ln384_bf16_kernel(const float* __restrict__ X,
                       const float* __restrict__ g,
                       const float* __restrict__ be,
                       unsigned short* __restrict__ Ypad,
                       int L) {
    const int wave = threadIdx.x >> 5;
    const int lane = threadIdx.x & 31;
    const int row  = blockIdx.x * 8 + wave;
    const int b    = row / L;
    const int t    = row - b * L;
    const float* xp = X + (long long)row * FF;
    float v[12], s = 0.f, sq = 0.f;
#pragma unroll
    for (int j = 0; j < 12; ++j) {
        v[j] = xp[lane + j * 32];
        s += v[j]; sq += v[j] * v[j];
    }
    for (int off = 16; off; off >>= 1) {
        s  += __shfl_xor(s,  off, 32);
        sq += __shfl_xor(sq, off, 32);
    }
    const float mu  = s * (1.0f / FF);
    const float var = sq * (1.0f / FF) - mu * mu;
    const float inv = rsqrtf(var + LN_EPS);
    unsigned short* yp = Ypad + ((long long)b * (L + 2) + 1 + t) * DD;
#pragma unroll
    for (int j = 0; j < 12; ++j) {
        const int c = lane + j * 32;
        yp[c] = f2bf((v[j] - mu) * inv * g[c] + be[c]);
    }
}

// ---------------------------------------------------------------------------
// fused LayerNorm + (F->1) linear + mask, wave-per-row
// ---------------------------------------------------------------------------
__global__ __launch_bounds__(256)
void ln_linear_kernel(const float* __restrict__ X,
                      const float* __restrict__ g,
                      const float* __restrict__ be,
                      const float* __restrict__ lw,
                      const float* __restrict__ lb,
                      const unsigned char* __restrict__ mask,
                      float* __restrict__ pred) {
    const int wave = threadIdx.x >> 5;
    const int lane = threadIdx.x & 31;
    const int row  = blockIdx.x * 8 + wave;
    const float* xp = X + (long long)row * FF;
    float v[12], s = 0.f, sq = 0.f;
#pragma unroll
    for (int j = 0; j < 12; ++j) {
        v[j] = xp[lane + j * 32];
        s += v[j]; sq += v[j] * v[j];
    }
    for (int off = 16; off; off >>= 1) {
        s  += __shfl_xor(s,  off, 32);
        sq += __shfl_xor(sq, off, 32);
    }
    const float mu  = s * (1.0f / FF);
    const float var = sq * (1.0f / FF) - mu * mu;
    const float inv = rsqrtf(var + LN_EPS);
    float d = 0.f;
#pragma unroll
    for (int j = 0; j < 12; ++j) {
        const int c = lane + j * 32;
        d += ((v[j] - mu) * inv * g[c] + be[c]) * lw[c];
    }
    for (int off = 16; off; off >>= 1) d += __shfl_xor(d, off, 32);
    if (lane == 0) pred[row] = mask[row] ? 0.0f : (d + lb[0]);
}

// ---------------------------------------------------------------------------
// final: out = xe + pitch_emb[bucket(pt)] + energy_emb[bucket(et)];
// also emits mel_mask passthrough as float
// ---------------------------------------------------------------------------
__device__ __forceinline__ int lower_bound255(const float* bins, float v) {
    int lo = 0, hi = NBINS - 1;                 // 255 bin edges
    while (lo < hi) { int mid = (lo + hi) >> 1; if (bins[mid] < v) lo = mid + 1; else hi = mid; }
    return lo;                                   // first i with bins[i] >= v (side='left')
}

__global__ void final_add_kernel(const float* __restrict__ xe,
                                 const float* __restrict__ pt,
                                 const float* __restrict__ et,
                                 const float* __restrict__ pemb,
                                 const float* __restrict__ eemb,
                                 const float* __restrict__ pbins,
                                 const float* __restrict__ ebins,
                                 const unsigned char* __restrict__ mmask,
                                 float* __restrict__ out,
                                 float* __restrict__ mask_out) {
    const int r = blockIdx.x;                    // (b*M + m)
    const int pi = lower_bound255(pbins, pt[r]);
    const int ei = lower_bound255(ebins, et[r]);
    const long long base = (long long)r * DD;
    for (int d = threadIdx.x; d < DD; d += blockDim.x)
        out[base + d] = xe[base + d] + pemb[pi * DD + d] + eemb[ei * DD + d];
    if (threadIdx.x == 0) mask_out[r] = mmask[r] ? 1.0f : 0.0f;
}

// ---------------------------------------------------------------------------
// host orchestration
// ---------------------------------------------------------------------------
extern "C" void kernel_launch(void* const* d_in, const int* in_sizes, int n_in,
                              void* d_out, int out_size, void* d_ws, size_t ws_size,
                              hipStream_t stream) {
    (void)in_sizes; (void)n_in; (void)out_size; (void)ws_size;

    // ---- inputs (setup_inputs() dict order, nested dicts in insertion order)
    const float*         x        = (const float*)d_in[0];
    const unsigned char* src_mask = (const unsigned char*)d_in[1];
    const unsigned char* mel_mask = (const unsigned char*)d_in[2];
    const int*           duration = (const int*)d_in[3];
    const float*         pitch_t  = (const float*)d_in[4];
    const float*         energy_t = (const float*)d_in[5];
    // d_in[6] = max_len scalar (compile-time MM)
    const float* P[3][10];   // dur / pitch / energy params: w1,b1,g1,be1,w2,b2,g2,be2,lw,lb
    for (int p = 0; p < 3; ++p)
        for (int j = 0; j < 10; ++j)
            P[p][j] = (const float*)d_in[7 + p * 10 + j];
    const float* pitch_emb  = (const float*)d_in[37];
    const float* energy_emb = (const float*)d_in[38];
    const float* pitch_bins = (const float*)d_in[39];
    const float* energy_bins= (const float*)d_in[40];

    // ---- output layout (flat concat in return order)
    float* out_base   = (float*)d_out;
    float* out_main   = out_base;                                   // [B,M,D]
    float* out_logdur = out_base + (long long)BB * MM * DD;         // [B,T]
    float* out_pitch  = out_logdur + (long long)BB * TT;            // [B,M]
    float* out_energy = out_pitch + (long long)BB * MM;             // [B,M]
    float* out_mellen = out_energy + (long long)BB * MM;            // [B]
    float* out_mask   = out_mellen + BB;                            // [B,M]

    // ---- workspace carve
    char* ws = (char*)d_ws;
    size_t off = 0;
    auto carve = [&](size_t bytes) { void* p = ws + off; off += (bytes + 255) & ~(size_t)255; return p; };
    int*            ends  = (int*)carve((size_t)BB * TT * 4);
    unsigned short* xbfp  = (unsigned short*)carve((size_t)BB * (TT + 2) * DD * 2); // padded
    float*          xe    = (float*)carve((size_t)BB * MM * DD * 4);
    unsigned short* xebfp = (unsigned short*)carve((size_t)BB * (MM + 2) * DD * 2); // padded
    unsigned short* wT[6];
    for (int i = 0; i < 6; ++i) wT[i] = (unsigned short*)carve((size_t)3 * DD * FF * 2);
    float*          Hbuf  = (float*)carve((size_t)BB * MM * FF * 4);
    unsigned short* A2p   = (unsigned short*)carve((size_t)BB * (MM + 2) * FF * 2); // padded

    // ---- prep
    zero_pad_rows_kernel<<<BB * 2, 128, 0, stream>>>(xbfp, TT);
    zero_pad_rows_kernel<<<BB * 2, 128, 0, stream>>>(xebfp, MM);
    {
        long long n = (long long)BB * TT * DD;
        f32_to_bf16_pad_kernel<<<(unsigned)((n + 255) / 256), 256, 0, stream>>>(x, xbfp, TT, n);
    }
    dur_scan_kernel<<<BB, TT, 0, stream>>>(duration, ends, out_mellen);
    length_reg_kernel<<<BB * MM, 128, 0, stream>>>(x, ends, xe, xebfp);
    {
        const int wn = 3 * DD * FF, wb = (wn + 255) / 256;
        wprep_kernel<<<wb, 256, 0, stream>>>(P[0][0], wT[0]);  // dur w1
        wprep_kernel<<<wb, 256, 0, stream>>>(P[0][4], wT[1]);  // dur w2
        wprep_kernel<<<wb, 256, 0, stream>>>(P[1][0], wT[2]);  // pitch w1
        wprep_kernel<<<wb, 256, 0, stream>>>(P[1][4], wT[3]);  // pitch w2
        wprep_kernel<<<wb, 256, 0, stream>>>(P[2][0], wT[4]);  // energy w1
        wprep_kernel<<<wb, 256, 0, stream>>>(P[2][4], wT[5]);  // energy w2
    }

    // ---- run one variance predictor (A2p halos zeroed per layout before use)
    auto run_predictor = [&](const unsigned short* Apad, int L, int nRows,
                             const float* const* p, const unsigned short* wt1,
                             const unsigned short* wt2, const unsigned char* mask,
                             float* pred) {
        const dim3 gb(nRows / 128, 2);
        const int lnb = nRows / 8;
        zero_pad_rows_kernel<<<BB * 2, 128, 0, stream>>>(A2p, L);
        conv_gemm_kernel<<<gb, 256, 0, stream>>>(Apad, wt1, p[1], Hbuf, L);
        ln384_bf16_kernel<<<lnb, 256, 0, stream>>>(Hbuf, p[2], p[3], A2p, L);
        conv_gemm_kernel<<<gb, 256, 0, stream>>>(A2p, wt2, p[5], Hbuf, L);
        ln_linear_kernel<<<lnb, 256, 0, stream>>>(Hbuf, p[6], p[7], p[8], p[9], mask, pred);
    };

    run_predictor(xbfp,  TT, BB * TT, P[0], wT[0], wT[1], src_mask, out_logdur);
    run_predictor(xebfp, MM, BB * MM, P[1], wT[2], wT[3], mel_mask, out_pitch);
    run_predictor(xebfp, MM, BB * MM, P[2], wT[4], wT[5], mel_mask, out_energy);

    // ---- final combine + passthrough outputs
    final_add_kernel<<<BB * MM, 128, 0, stream>>>(xe, pitch_t, energy_t,
                                                  pitch_emb, energy_emb,
                                                  pitch_bins, energy_bins,
                                                  mel_mask, out_main, out_mask);
}
